// Model_9972914061942
// MI455X (gfx1250) — matrix operand order repaired and checked by reading, not compiled, not run on hardware
//
#include <hip/hip_runtime.h>
#include <hip/hip_bf16.h>
#include <stdint.h>

typedef _Float16 h16;
typedef __attribute__((ext_vector_type(16))) _Float16 v16h;
typedef __attribute__((ext_vector_type(8)))  _Float16 v8h;
typedef __attribute__((ext_vector_type(8)))  float    v8f;
typedef __attribute__((ext_vector_type(4)))  unsigned u32x4;
typedef __attribute__((ext_vector_type(8)))  int      i32x8;
typedef __attribute__((ext_vector_type(4)))  int      i32x4;
typedef __attribute__((address_space(1))) i32x4 as1_i32x4;
typedef __attribute__((address_space(3))) i32x4 as3_i32x4;

#define N_SAMP 32768
#define CH     128
#define NATOMS 1024
#define ASIZE  512
#define BATCH  2
#define KSP    512
#define NROOMS 8
#define NP     (N_SAMP + 32)     // padded time axis for f16 encoded (offset +16)
#define CAP    2048              // sparse-event list capacity per batch
#define NBINS  2048

#define HAS_ASYNC_LDS   __has_builtin(__builtin_amdgcn_global_load_async_to_lds_b128)
#define HAS_WAIT_ASYNC  __has_builtin(__builtin_amdgcn_s_wait_asynccnt)
#define HAS_TDM         __has_builtin(__builtin_amdgcn_tensor_load_to_lds)
#define HAS_WAIT_TENSOR __has_builtin(__builtin_amdgcn_s_wait_tensorcnt)

// ---------------------------------------------------------------- utilities
__global__ void k_zero_u32(unsigned* p, int n) {
  int i = blockIdx.x * 256 + threadIdx.x;
  if (i < n) p[i] = 0u;
}

// ------------------------------------------------- front conv: x -> h (curA)
__global__ void k_front(const float* __restrict__ x, const float* __restrict__ cw,
                        const float* __restrict__ cb, float* __restrict__ cur) {
  int i = blockIdx.x * 256 + threadIdx.x;
  if (i >= BATCH * CH * N_SAMP) return;
  int t = i % N_SAMP;
  int c = (i / N_SAMP) % CH;
  int b = i / (N_SAMP * CH);
  const float* xb = x + (size_t)b * N_SAMP;
  float s = cb[c];
#pragma unroll
  for (int k = 0; k < 7; ++k) {
    int tt = t + k - 3;
    if (tt >= 0 && tt < N_SAMP) s += cw[c * 7 + k] * xb[tt];
  }
  cur[i] = s;
}

// --------------------------------------- one dilated layer (WMMA f16 GEMM)
// y[co,t] = lrelu( sum_ci W0[co,ci]*cur[ci,t] + W1[co,ci]*cur[ci,t+dil] + b )
// K = k*128 + ci (256). Block tile: 64 co x 32 t, 8 waves of 16x16.
__global__ __launch_bounds__(256) void k_dilated(
    const float* __restrict__ curIn, float* __restrict__ curOut,
    float* __restrict__ acc, const float* __restrict__ w,   // [co][ci][2]
    const float* __restrict__ bias, int dil, int first) {
  __shared__ __align__(16) h16 wlds[64][256];
  __shared__ __align__(16) h16 btile[64][136];
  const int tid = threadIdx.x;
  const int b = blockIdx.z;
  const int coBase = blockIdx.y * 64;
  const int t0 = blockIdx.x * 32;
  const float* curB = curIn + (size_t)b * CH * N_SAMP;

  for (int e = tid; e < 64 * 256; e += 256) {
    int co = e >> 8, K = e & 255;
    int k = K >> 7, ci = K & 127;
    wlds[co][K] = (h16)w[((size_t)(coBase + co) * CH + ci) * 2 + k];
  }
  {
    int tl = tid & 31;
    for (int ci = tid >> 5; ci < CH; ci += 8) {
      const float* row = curB + (size_t)ci * N_SAMP;
      btile[tl][ci] = (h16)row[t0 + tl];
      int t2 = t0 + dil + tl;
      btile[32 + tl][ci] = (t2 < N_SAMP) ? (h16)row[t2] : (h16)0.f;
    }
  }
  __syncthreads();

  const int wv = tid >> 5, lane = tid & 31, l16 = lane & 15;
  const int mSub = (wv >> 1) * 16;
  const int nSub = (wv & 1) * 16;
  v8f cacc = {};
#pragma unroll
  for (int kk = 0; kk < 256; kk += 32) {
    int a1 = kk + ((lane < 16) ? 0 : 8);
    v8h alo = *(const v8h*)&wlds[mSub + l16][a1];
    v8h ahi = *(const v8h*)&wlds[mSub + l16][a1 + 16];
    int cb0 = (kk & 127) + ((lane < 16) ? 0 : 8);
    int rowb = nSub + l16 + ((kk >> 7) ? 32 : 0);
    v8h blo = *(const v8h*)&btile[rowb][cb0];
    v8h bhi = *(const v8h*)&btile[rowb][cb0 + 16];
    v16h A, Bm;
#pragma unroll
    for (int q = 0; q < 8; ++q) { A[q] = alo[q]; A[q + 8] = ahi[q]; Bm[q] = blo[q]; Bm[q + 8] = bhi[q]; }
    cacc = __builtin_amdgcn_wmma_f32_16x16x32_f16(false, A, false, Bm, (short)0, cacc, false, false);
  }

  int t = t0 + nSub + l16;
#pragma unroll
  for (int v = 0; v < 8; ++v) {
    int co = coBase + mSub + v + ((lane < 16) ? 0 : 8);
    float y = cacc[v] + bias[co];
    y = (y > 0.f) ? y : 0.2f * y;
    size_t idx = ((size_t)b * CH + co) * N_SAMP + t;
    curOut[idx] = curIn[idx] + y;
    acc[idx] = first ? y : (acc[idx] + y);
  }
}

// -------------------------------------------------- summary = mean_t(acc)
__global__ void k_summary(const float* __restrict__ acc, float* __restrict__ summary) {
  __shared__ float red[256];
  int row = blockIdx.x;                 // b*CH + c
  const float* p = acc + (size_t)row * N_SAMP;
  float s = 0.f;
  for (int t = threadIdx.x; t < N_SAMP; t += 256) s += p[t];
  red[threadIdx.x] = s; __syncthreads();
  for (int st = 128; st; st >>= 1) { if (threadIdx.x < st) red[threadIdx.x] += red[threadIdx.x + st]; __syncthreads(); }
  if (!threadIdx.x) summary[row] = red[0] * (1.f / N_SAMP);
}

// ------------- encoded f32 [b][ci][t] -> padded f16 TIME-MAJOR [b][t][ci]
__global__ void k_enc16(const float* __restrict__ acc, h16* __restrict__ encp2) {
  size_t i = (size_t)blockIdx.x * 256 + threadIdx.x;
  if (i >= (size_t)BATCH * NP * CH) return;
  int ci = (int)(i % CH);
  int j  = (int)((i / CH) % NP);
  int b  = (int)(i / ((size_t)NP * CH));
  int t = j - 16;
  encp2[i] = (t >= 0 && t < N_SAMP) ? (h16)acc[((size_t)b * CH + ci) * N_SAMP + t] : (h16)0.f;
}

// ------------------------------ atom weights f32[atom][ci][7] -> f16 K=k*128+ci
__global__ void k_cvt_aw(const float* __restrict__ aw, h16* __restrict__ awf) {
  int i = blockIdx.x * 256 + threadIdx.x;
  if (i >= NATOMS * 896) return;
  int K = i % 896, atom = i / 896;
  int k = K >> 7, ci = K & 127;
  awf[i] = (h16)aw[(atom * CH + ci) * 7 + k];
}

// ------------------------- atom conv (WMMA): dense a -> d_out sparse region
// B tile staged via ASYNCcnt global->LDS copies; A tile staged per 128-K block
// via the Tensor Data Mover (TENSORcnt), one wave issuing the descriptor.
__global__ __launch_bounds__(256) void k_atom(
    const h16* __restrict__ awf, const h16* __restrict__ encp2,
    const float* __restrict__ ab, float* __restrict__ dense) {
  __shared__ __align__(16) h16 btile[38][128];   // [t0-3+j][ci], rows contiguous
  __shared__ __align__(16) h16 atile[64][128];   // 64 atoms x current K-block
  const int tid = threadIdx.x;
  const int b = blockIdx.z;
  const int atomBase = blockIdx.y * 64;
  const int t0 = blockIdx.x * 32;
  const h16* encb = encp2 + (size_t)b * NP * CH + (size_t)(16 + t0 - 3) * CH;

  // ---- stage B tile: 38 rows x 256B contiguous; 304 x 16B lane-transfers
  for (int e = tid; e < 38 * 8; e += 256) {
    int j = e >> 3, c = e & 7;
    const h16* g = encb + j * CH + c * 8;
    h16* l = &btile[j][c * 8];
#if HAS_ASYNC_LDS
    __builtin_amdgcn_global_load_async_to_lds_b128(
        (as1_i32x4*)(unsigned long long)(uintptr_t)g,
        (as3_i32x4*)(unsigned)(uintptr_t)l, 0, 0);
#else
    *(v8h*)l = *(const v8h*)g;
#endif
  }
#if HAS_ASYNC_LDS && HAS_WAIT_ASYNC
  __builtin_amdgcn_s_wait_asynccnt(0);
#endif
  __syncthreads();

  const int wv = tid >> 5, lane = tid & 31, l16 = lane & 15;
  const int mSub = (wv >> 1) * 16, nSub = (wv & 1) * 16;
  v8f cacc = {};

#if HAS_TDM
  for (int kb = 0; kb < 7; ++kb) {
    __syncthreads();                       // atile consumed by all waves
    if (tid < 32) {                        // one wave drives the TDM
      const h16* gsrc = awf + (size_t)atomBase * 896 + kb * 128;
      unsigned long long ga = (unsigned long long)(uintptr_t)gsrc;
      unsigned ldsa = (unsigned)(uintptr_t)&atile[0][0];
      u32x4 g0;
      g0[0] = 1u;                                           // count=1, user desc
      g0[1] = ldsa;                                         // lds_addr (bytes)
      g0[2] = (unsigned)(ga & 0xffffffffu);                 // global_addr[31:0]
      g0[3] = (unsigned)((ga >> 32) & 0x01ffffffu) | (2u << 30); // [56:32] | type=2
      i32x8 g1;
      g1[0] = (int)(1u << 16);                              // data_size=2B
      g1[1] = (int)((896u & 0xffffu) << 16);                // tensor_dim0 lo16
      g1[2] = (int)((896u >> 16) | (1024u << 16));          // dim0 hi | tensor_dim1 lo
      g1[3] = (int)(128u << 16);                            // dim1 hi | tile_dim0=128
      g1[4] = (int)64u;                                     // tile_dim1=64, tile_dim2=0
      g1[5] = (int)896u;                                    // tensor_dim0_stride lo32
      g1[6] = 0;                                            // stride hi | dim1_stride lo
      g1[7] = 0;
      i32x4 gz; gz[0] = 0; gz[1] = 0; gz[2] = 0; gz[3] = 0;
#if defined(__clang_major__) && __clang_major__ >= 23
      i32x8 gz8; for (int q = 0; q < 8; ++q) gz8[q] = 0;
      __builtin_amdgcn_tensor_load_to_lds(g0, g1, gz, gz, gz8, 0);
#else
      __builtin_amdgcn_tensor_load_to_lds(g0, g1, gz, gz, 0);
#endif
#if HAS_WAIT_TENSOR
      __builtin_amdgcn_s_wait_tensorcnt(0);
#endif
    }
    __syncthreads();
#pragma unroll
    for (int k2 = 0; k2 < 128; k2 += 32) {
      int a1 = k2 + ((lane < 16) ? 0 : 8);
      v8h alo = *(const v8h*)&atile[mSub + l16][a1];
      v8h ahi = *(const v8h*)&atile[mSub + l16][a1 + 16];
      int cb0 = k2 + ((lane < 16) ? 0 : 8);
      int rowb = nSub + l16 + kb;
      v8h blo = *(const v8h*)&btile[rowb][cb0];
      v8h bhi = *(const v8h*)&btile[rowb][cb0 + 16];
      v16h A, Bm;
#pragma unroll
      for (int q = 0; q < 8; ++q) { A[q] = alo[q]; A[q + 8] = ahi[q]; Bm[q] = blo[q]; Bm[q + 8] = bhi[q]; }
      cacc = __builtin_amdgcn_wmma_f32_16x16x32_f16(false, A, false, Bm, (short)0, cacc, false, false);
    }
  }
#else
  // fallback: A operand streamed from global per lane
  const h16* arow = awf + (size_t)(atomBase + mSub + l16) * 896;
#pragma unroll 4
  for (int kk = 0; kk < 896; kk += 32) {
    int a1 = kk + ((lane < 16) ? 0 : 8);
    __builtin_prefetch(arow + a1 + 64, 0, 1);
    v8h alo = *(const v8h*)(arow + a1);
    v8h ahi = *(const v8h*)(arow + a1 + 16);
    int cb0 = (kk & 127) + ((lane < 16) ? 0 : 8);
    int rowb = nSub + l16 + (kk >> 7);
    v8h blo = *(const v8h*)&btile[rowb][cb0];
    v8h bhi = *(const v8h*)&btile[rowb][cb0 + 16];
    v16h A, Bm;
#pragma unroll
    for (int q = 0; q < 8; ++q) { A[q] = alo[q]; A[q + 8] = ahi[q]; Bm[q] = blo[q]; Bm[q + 8] = bhi[q]; }
    cacc = __builtin_amdgcn_wmma_f32_16x16x32_f16(false, A, false, Bm, (short)0, cacc, false, false);
  }
#endif

  int t = t0 + nSub + l16;
#pragma unroll
  for (int v = 0; v < 8; ++v) {
    int atom = atomBase + mSub + v + ((lane < 16) ? 0 : 8);
    dense[((size_t)b * NATOMS + atom) * N_SAMP + t] = cacc[v] + ab[atom];
  }
}

// ---------------------------------------- top-k via exponent-bucket histogram
__global__ void k_hist(const float* __restrict__ dense, unsigned* __restrict__ hist) {
  size_t i = (size_t)blockIdx.x * 256 + threadIdx.x;
  const size_t per = (size_t)NATOMS * N_SAMP;
  if (i >= (size_t)BATCH * per) return;
  int b = (int)(i / per);
  float v = dense[i];
  if (v > 0.f) {
    unsigned bin = __float_as_uint(v) >> 20;     // positive floats -> [0,2048)
    atomicAdd(&hist[b * NBINS + bin], 1u);
  }
}

__global__ void k_thresh(const unsigned* __restrict__ hist, float* __restrict__ thr) {
  if (threadIdx.x) return;
  int b = blockIdx.x;
  unsigned cum = 0; int bin = NBINS - 1;
  for (; bin >= 0; --bin) { cum += hist[b * NBINS + bin]; if (cum >= KSP) break; }
  if (bin < 0) bin = 0;
  thr[b] = __uint_as_float((unsigned)bin << 20);
}

__global__ void k_select(float* __restrict__ dense, const float* __restrict__ thr,
                         unsigned* __restrict__ cnt, unsigned* __restrict__ eidx,
                         float* __restrict__ eval) {
  size_t i = (size_t)blockIdx.x * 256 + threadIdx.x;
  const size_t per = (size_t)NATOMS * N_SAMP;
  if (i >= (size_t)BATCH * per) return;
  int b = (int)(i / per);
  float v = dense[i];
  bool keep = v > thr[b];
  dense[i] = keep ? v : 0.f;                       // relu(sparsify) in place
  if (keep) {
    unsigned slot = atomicAdd(&cnt[b], 1u);
    if (slot < CAP) {
      eidx[b * CAP + slot] = (unsigned)(i - (size_t)b * per);
      eval[b * CAP + slot] = v;
    }
  }
}

// ------------------------------------------------- unit-norm atoms (d)
__global__ void k_norm_atoms(const float* __restrict__ d, float* __restrict__ atomN) {
  __shared__ float red[256];
  int atom = blockIdx.x;
  const float* p = d + (size_t)atom * ASIZE;
  float s = 0.f;
  for (int i = threadIdx.x; i < ASIZE; i += 256) { float x = p[i]; s += x * x; }
  red[threadIdx.x] = s; __syncthreads();
  for (int st = 128; st; st >>= 1) { if (threadIdx.x < st) red[threadIdx.x] += red[threadIdx.x + st]; __syncthreads(); }
  float inv = 1.f / (sqrtf(red[0]) + 1e-8f);
  for (int i = threadIdx.x; i < ASIZE; i += 256) atomN[(size_t)atom * ASIZE + i] = p[i] * inv;
}

// ----------------- sparse event scatter == sum_atoms fft_convolve(a, atoms)
__global__ void k_scatter(const unsigned* __restrict__ cnt, const unsigned* __restrict__ eidx,
                          const float* __restrict__ eval, const float* __restrict__ atomN,
                          float* __restrict__ finalb) {
  int b = blockIdx.y;
  unsigned n = cnt[b]; if (n > CAP) n = CAP;
  unsigned e = blockIdx.x;
  if (e >= n) return;
  unsigned fi = eidx[b * CAP + e];
  float v = eval[b * CAP + e];
  int atom = fi / N_SAMP, pos = fi % N_SAMP;
  for (int tap = threadIdx.x; tap < ASIZE; tap += 256) {
    int t = pos + tap;
    if (t < N_SAMP) atomicAdd(&finalb[(size_t)b * N_SAMP + t], v * atomN[(size_t)atom * ASIZE + tap]);
  }
}

// --------------------------- 3-layer MLP + LN + room argmax + mix softmax
__global__ __launch_bounds__(128) void k_mlp(
    const float* __restrict__ summary, const float* __restrict__ mlp_w,
    const float* __restrict__ mlp_b, const float* __restrict__ ln_w,
    const float* __restrict__ ln_b, const float* __restrict__ room_w,
    const float* __restrict__ room_b, const float* __restrict__ mix_w,
    const float* __restrict__ mix_b, int* __restrict__ roomIdx, float* __restrict__ mixOut) {
  __shared__ float cS[CH], red[128];
  int b = blockIdx.x, j = threadIdx.x;
  cS[j] = summary[b * CH + j];
  __syncthreads();
  for (int l = 0; l < 3; ++l) {
    const float* W = mlp_w + l * CH * CH;
    float s = mlp_b[l * CH + j];
    for (int i = 0; i < CH; ++i) s += cS[i] * W[i * CH + j];
    s = (s > 0.f) ? s : 0.2f * s;
    __syncthreads();
    red[j] = s; __syncthreads();
    for (int st = 64; st; st >>= 1) { if (j < st) red[j] += red[j + st]; __syncthreads(); }
    float mu = red[0] * (1.f / CH); __syncthreads();
    float dd = s - mu;
    red[j] = dd * dd; __syncthreads();
    for (int st = 64; st; st >>= 1) { if (j < st) red[j] += red[j + st]; __syncthreads(); }
    float var = red[0] * (1.f / CH); __syncthreads();
    cS[j] = dd * rsqrtf(var + 1e-5f) * ln_w[l * CH + j] + ln_b[l * CH + j];
    __syncthreads();
  }
  if (j < NROOMS) {
    float s = room_b[j];
    for (int i = 0; i < CH; ++i) s += cS[i] * room_w[i * NROOMS + j];
    red[j] = s;
  }
  if (j >= 64 && j < 66) {
    int m = j - 64;
    float s = mix_b[m];
    for (int i = 0; i < CH; ++i) s += cS[i] * mix_w[i * 2 + m];
    red[32 + m] = s;
  }
  __syncthreads();
  if (j == 0) {
    int best = 0; float bv = red[0];
    for (int r = 1; r < NROOMS; ++r) if (red[r] > bv) { bv = red[r]; best = r; }
    roomIdx[b] = best;
    float m0 = red[32], m1 = red[33];
    float mx = fmaxf(m0, m1);
    float e0 = __expf(m0 - mx), e1 = __expf(m1 - mx);
    mixOut[b * 2 + 0] = e0 / (e0 + e1);
    mixOut[b * 2 + 1] = e1 / (e0 + e1);
  }
}

// ------------------- wet = causal linear conv of final with selected room
__global__ __launch_bounds__(256) void k_wet(
    const float* __restrict__ finalb, const float* __restrict__ rooms,
    const int* __restrict__ roomIdx, float* __restrict__ wet) {
  __shared__ float fS[256];
  __shared__ float rS[512];
  int b = blockIdx.y, t0 = blockIdx.x * 256, tl = threadIdx.x;
  const float* room = rooms + (size_t)roomIdx[b] * N_SAMP;
  const float* fb = finalb + (size_t)b * N_SAMP;
  float s = 0.f;
  for (int u0 = 0; u0 <= t0; u0 += 256) {
    fS[tl] = fb[u0 + tl];
    int d = t0 - u0, m0 = d - 255;
    int i1 = m0 + tl;        rS[tl]       = (i1 >= 0)                ? room[i1] : 0.f;
    int i2 = m0 + 256 + tl;  rS[256 + tl] = (i2 >= 0 && i2 < N_SAMP) ? room[i2] : 0.f;
    __syncthreads();
#pragma unroll 8
    for (int jj = 0; jj < 256; ++jj) s += fS[jj] * rS[tl - jj + 255];
    __syncthreads();
  }
  wet[(size_t)b * N_SAMP + t0 + tl] = s;
}

__global__ void k_mixout(const float* __restrict__ finalb, const float* __restrict__ wet,
                         const float* __restrict__ mix, float* __restrict__ out) {
  int i = blockIdx.x * 256 + threadIdx.x;
  if (i >= BATCH * N_SAMP) return;
  int b = i / N_SAMP;
  out[i] = finalb[i] * mix[b * 2] + wet[i] * mix[b * 2 + 1];
}

// ---------------------------------------------------------------- launcher
extern "C" void kernel_launch(void* const* d_in, const int* in_sizes, int n_in,
                              void* d_out, int out_size, void* d_ws, size_t ws_size,
                              hipStream_t stream) {
  const float* x      = (const float*)d_in[0];
  const float* cw     = (const float*)d_in[1];
  const float* cb     = (const float*)d_in[2];
  const float* enc_w  = (const float*)d_in[3];
  const float* enc_b  = (const float*)d_in[4];
  const float* aw     = (const float*)d_in[5];
  const float* ab     = (const float*)d_in[6];
  const float* dmat   = (const float*)d_in[7];
  const float* mlp_w  = (const float*)d_in[8];
  const float* mlp_b  = (const float*)d_in[9];
  const float* ln_w   = (const float*)d_in[10];
  const float* ln_b   = (const float*)d_in[11];
  const float* room_w = (const float*)d_in[12];
  const float* room_b = (const float*)d_in[13];
  const float* mix_w  = (const float*)d_in[14];
  const float* mix_b  = (const float*)d_in[15];
  const float* rooms  = (const float*)d_in[16];

  float* out_main = (float*)d_out;                  // [B,1,N]
  float* dense    = (float*)d_out + BATCH * N_SAMP; // [B,1024,N] sparse_features

  const size_t BCHN = (size_t)BATCH * CH * N_SAMP;
  char* base = (char*)d_ws;
  size_t off = 0;
  auto take = [&](size_t bytes) { size_t r = off; off = (off + bytes + 255) & ~(size_t)255; return r; };
  float*    curA    = (float*)(base + take(BCHN * 4));
  float*    curB    = (float*)(base + take(BCHN * 4));
  float*    acc     = (float*)(base + take(BCHN * 4));
  h16*      encp2   = (h16*)  (base + take((size_t)BATCH * NP * CH * 2));
  h16*      awf     = (h16*)  (base + take((size_t)NATOMS * 896 * 2));
  float*    atomN   = (float*)(base + take((size_t)NATOMS * ASIZE * 4));
  float*    finalb  = (float*)(base + take((size_t)BATCH * N_SAMP * 4));
  float*    wet     = (float*)(base + take((size_t)BATCH * N_SAMP * 4));
  float*    summary = (float*)(base + take((size_t)BATCH * CH * 4));
  float*    thr     = (float*)(base + take(BATCH * 4));
  unsigned* hist    = (unsigned*)(base + take((size_t)BATCH * NBINS * 4));
  unsigned* cnt     = (unsigned*)(base + take(BATCH * 4));
  unsigned* eidx    = (unsigned*)(base + take((size_t)BATCH * CAP * 4));
  float*    evals   = (float*)(base + take((size_t)BATCH * CAP * 4));
  int*      roomIdx = (int*)  (base + take(BATCH * 4));
  float*    mix     = (float*)(base + take(BATCH * 2 * 4));
  (void)ws_size; (void)n_in; (void)in_sizes; (void)out_size;

  k_zero_u32<<<(BATCH * N_SAMP + 255) / 256, 256, 0, stream>>>((unsigned*)finalb, BATCH * N_SAMP);
  k_zero_u32<<<(BATCH * NBINS + 255) / 256, 256, 0, stream>>>(hist, BATCH * NBINS);
  k_zero_u32<<<1, 256, 0, stream>>>(cnt, BATCH);

  k_front<<<(int)((BCHN + 255) / 256), 256, 0, stream>>>(x, cw, cb, curA);

  const int dils[12] = {1, 2, 4, 8, 16, 32, 64, 128, 256, 512, 1024, 1};
  dim3 gD(N_SAMP / 32, CH / 64, BATCH);
  for (int l = 0; l < 12; ++l) {
    const float* ci = (l & 1) ? curB : curA;
    float*       co = (l & 1) ? curA : curB;
    k_dilated<<<gD, 256, 0, stream>>>(ci, co, acc, enc_w + (size_t)l * CH * CH * 2,
                                      enc_b + (size_t)l * CH, dils[l], l == 0);
  }

  k_summary<<<BATCH * CH, 256, 0, stream>>>(acc, summary);
  k_enc16<<<(int)(((size_t)BATCH * NP * CH + 255) / 256), 256, 0, stream>>>(acc, encp2);
  k_cvt_aw<<<(NATOMS * 896 + 255) / 256, 256, 0, stream>>>(aw, awf);

  dim3 gA(N_SAMP / 32, NATOMS / 64, BATCH);
  k_atom<<<gA, 256, 0, stream>>>(awf, encp2, ab, dense);

  int selBlocks = (int)(((size_t)BATCH * NATOMS * N_SAMP + 255) / 256);
  k_hist<<<selBlocks, 256, 0, stream>>>(dense, hist);
  k_thresh<<<BATCH, 64, 0, stream>>>(hist, thr);
  k_select<<<selBlocks, 256, 0, stream>>>(dense, thr, cnt, eidx, evals);

  k_norm_atoms<<<NATOMS, 256, 0, stream>>>(dmat, atomN);
  dim3 gS(CAP, BATCH);
  k_scatter<<<gS, 256, 0, stream>>>(cnt, eidx, evals, atomN, finalb);

  k_mlp<<<BATCH, 128, 0, stream>>>(summary, mlp_w, mlp_b, ln_w, ln_b,
                                   room_w, room_b, mix_w, mix_b, roomIdx, mix);
  dim3 gW(N_SAMP / 256, BATCH);
  k_wet<<<gW, 256, 0, stream>>>(finalb, rooms, roomIdx, wet);
  k_mixout<<<(BATCH * N_SAMP + 255) / 256, 256, 0, stream>>>(finalb, wet, mix, out_main);
}
